// Attention_87978110091695
// MI455X (gfx1250) — matrix operand order repaired and checked by reading, not compiled, not run on hardware
//
#include <hip/hip_runtime.h>

// ---------------------------------------------------------------------------
// Causal attention forward for B=2,H=16,S=2048,D=64 (fp32 in, fp32 out).
// Outputs (concatenated in d_out): out [B,H,S,D] then attn_probs [B,H,S,S].
//
// Strategy: 128-thread block (4 wave32) covers 64 consecutive query rows;
// each wave owns a 16-query tile. K/V 32-key chunks are staged (fp32->bf16)
// into LDS once per block and shared by all 4 waves. QK^T and P*V use
// v_wmma_f32_16x16x32_bf16; softmax is fp32 with __shfl_xor row reductions.
// Two passes: (1) online softmax stats, (2) recompute scores, write
// normalized P to global, and accumulate O = P*V via an LDS bf16 round-trip.
// All fragment loads use distinct registers so the scheduler can issue them
// back-to-back with partial s_wait_dscnt; V fragments are hoisted ahead of
// the P/exp computation to hide LDS latency. __builtin_prefetch pulls the
// next chunk (global_prefetch) during compute.
// ---------------------------------------------------------------------------

constexpr int B = 2, H = 16, S = 2048, D = 64;
constexpr float SCALE = 0.125f;          // 1/sqrt(64)
constexpr float NEG_INF_MASK = -10000.0f;

typedef __attribute__((ext_vector_type(16))) __bf16 v16bf;
typedef __attribute__((ext_vector_type(8)))  float  v8f;

union FragBF {
    v16bf v;
    unsigned short h[16];
    unsigned int   w[8];
};

__device__ __forceinline__ unsigned short f32_to_bf16(float x) {
    __bf16 b = (__bf16)x;                       // hardware cvt path
    return __builtin_bit_cast(unsigned short, b);
}
__device__ __forceinline__ unsigned int pack_bf16x2(float x, float y) {
    return (unsigned int)f32_to_bf16(x) | ((unsigned int)f32_to_bf16(y) << 16);
}

__device__ __forceinline__ float redmax16(float x) {
#pragma unroll
    for (int off = 1; off < 16; off <<= 1)
        x = fmaxf(x, __shfl_xor(x, off, 32));
    return x;
}
__device__ __forceinline__ float redsum16(float x) {
#pragma unroll
    for (int off = 1; off < 16; off <<= 1)
        x += __shfl_xor(x, off, 32);
    return x;
}

// Stage a 32x64 fp32 tile -> bf16 LDS tile with all 128 threads.
__device__ __forceinline__ void stage32x64_blk(const float* __restrict__ src,
                                               unsigned short* dst, int tid) {
    const int row = tid >> 2, qr = tid & 3;     // 4 threads per 64-wide row
    const float4* s = (const float4*)(src + row * 64 + qr * 16);
    unsigned int* d = (unsigned int*)(dst + row * 64 + qr * 16);
#pragma unroll
    for (int j = 0; j < 4; ++j) {
        float4 f = s[j];
        d[2 * j]     = pack_bf16x2(f.x, f.y);
        d[2 * j + 1] = pack_bf16x2(f.z, f.w);
    }
}

// Stage a 64x64 fp32 tile (Q for the whole block) -> bf16 LDS tile.
__device__ __forceinline__ void stage64x64_blk(const float* __restrict__ src,
                                               unsigned short* dst, int tid) {
    const int row = tid >> 1, hr = tid & 1;     // 2 threads per row
    const float4* s = (const float4*)(src + row * 64 + hr * 32);
    unsigned int* d = (unsigned int*)(dst + row * 64 + hr * 32);
#pragma unroll
    for (int j = 0; j < 8; ++j) {
        float4 f = s[j];
        d[2 * j]     = pack_bf16x2(f.x, f.y);
        d[2 * j + 1] = pack_bf16x2(f.z, f.w);
    }
}

// A-fragment (16x32 bf16) per ISA 7.12.2: lane m=lane&15 holds row m;
// half=lane>>4 selects K groups {0..7,16..23} vs {8..15,24..31}.
__device__ __forceinline__ void load_afrag(const unsigned short* base, int rstride,
                                           int lane, FragBF& f) {
    const int m = lane & 15, hfl = lane >> 4;
    const unsigned short* r = base + m * rstride;
#pragma unroll
    for (int p = 0; p < 8; ++p) {
        const int col = ((p < 4) ? 2 * p : 2 * p + 8) + hfl * 8;  // even -> 4B aligned
        f.w[p] = *(const unsigned int*)(r + col);
    }
}

// B-fragment for QK^T: B[k=d][n=key]; staged tile is keys(rows) x d(cols).
__device__ __forceinline__ void load_bfrag_k(const unsigned short* tile, int row0,
                                             int colbase, int lane, FragBF& f) {
    const int nn = lane & 15, hfl = lane >> 4;
    const unsigned int* r =
        (const unsigned int*)(tile + (row0 + nn) * 64 + colbase);
#pragma unroll
    for (int p = 0; p < 8; ++p) f.w[p] = r[((p < 4) ? p : p + 4) + 4 * hfl];
}

// B-fragment for P*V: B[k=key][n=d]; staged V tile is keys(rows) x d(cols).
__device__ __forceinline__ void load_bfrag_v(const unsigned short* tile, int dcol,
                                             int lane, FragBF& f) {
    const int nn = lane & 15, hfl = lane >> 4;
#pragma unroll
    for (int i = 0; i < 16; ++i)
        f.h[i] = tile[(((i < 8) ? i : i + 8) + 8 * hfl) * 64 + dcol + nn];
}

// Compute the 16x32 score block (two 16x16 C tiles) for one key chunk.
// Distinct fragment registers -> loads issue back-to-back, partial dscnt waits.
__device__ __forceinline__ void score_chunk(const FragBF& aq0, const FragBF& aq1,
                                            const unsigned short* lk, int lane,
                                            v8f& t0, v8f& t1) {
    FragBF bk0, bk1, bk2, bk3;
    load_bfrag_k(lk, 0, 0, lane, bk0);
    load_bfrag_k(lk, 0, 32, lane, bk1);
    load_bfrag_k(lk, 16, 0, lane, bk2);
    load_bfrag_k(lk, 16, 32, lane, bk3);
    const v8f zero = {};
    t0 = __builtin_amdgcn_wmma_f32_16x16x32_bf16(false, aq0.v, false, bk0.v,
                                                 (short)0, zero, false, false);
    t0 = __builtin_amdgcn_wmma_f32_16x16x32_bf16(false, aq1.v, false, bk1.v,
                                                 (short)0, t0, false, false);
    t1 = __builtin_amdgcn_wmma_f32_16x16x32_bf16(false, aq0.v, false, bk2.v,
                                                 (short)0, zero, false, false);
    t1 = __builtin_amdgcn_wmma_f32_16x16x32_bf16(false, aq1.v, false, bk3.v,
                                                 (short)0, t1, false, false);
}

__global__ __launch_bounds__(128)
void fa_fwd_kernel(const float* __restrict__ q, const float* __restrict__ k,
                   const float* __restrict__ v, float* __restrict__ out,
                   float* __restrict__ attn) {
    __shared__ unsigned short lds_q[64 * 64];       // 4 query tiles (bf16)
    __shared__ unsigned short lds_k[32 * 64];       // shared K chunk
    __shared__ unsigned short lds_v[32 * 64];       // shared V chunk
    __shared__ unsigned short lds_p[4 * 16 * 32];   // per-wave P tile

    const int tid  = threadIdx.x;
    const int wave = tid >> 5;
    const int lane = tid & 31;
    const int hfl = lane >> 4;        // C-layout: rows r + 8*hfl
    const int nn  = lane & 15;        // C-layout: column
    const int base = blockIdx.x * 64; // first query row of this block
    const int q0 = base + wave * 16;  // this wave's query tile
    const long bh = (long)blockIdx.z * H + blockIdx.y;

    const float* qp = q + bh * S * D;
    const float* kp = k + bh * S * D;
    const float* vp = v + bh * S * D;
    float* op = out + bh * S * D;
    float* ap = attn + bh * (long)S * S;

    // Uniform chunk count for the whole block: keys [0, base+64).
    // Early waves' extra chunks are fully masked -> exact zeros (correct).
    const int nchunks = base / 32 + 2;
    const int kend = base + 64;
    unsigned short* myp = lds_p + wave * (16 * 32);

    // ---- Q tiles -> per-wave A-fragments (d 0..31 and 32..63) ----
    stage64x64_blk(qp + base * 64, lds_q, tid);
    __syncthreads();
    FragBF aq0, aq1;
    load_afrag(lds_q + (wave * 16) * 64 + 0, 64, lane, aq0);
    load_afrag(lds_q + (wave * 16) * 64 + 32, 64, lane, aq1);

    // ---- pass 1: online softmax stats (m, l) per row ----
    float mrow[8], lrow[8];
#pragma unroll
    for (int r = 0; r < 8; ++r) { mrow[r] = -3.0e38f; lrow[r] = 0.0f; }

    for (int ch = 0; ch < nchunks; ++ch) {
        const int kbase = ch * 32;
        __syncthreads();
        stage32x64_blk(kp + kbase * 64, lds_k, tid);
        __syncthreads();
        if (ch + 1 < nchunks)  // overlap next chunk's fetch with compute
            __builtin_prefetch(kp + (kbase + 32) * 64 + tid * 16, 0, 1);

        v8f t0, t1;
        score_chunk(aq0, aq1, lds_k, lane, t0, t1);

#pragma unroll
        for (int r = 0; r < 8; ++r) {
            const int row = q0 + r + 8 * hfl;
            float s0 = t0[r] * SCALE, s1 = t1[r] * SCALE;
            if (kbase + nn > row)      s0 = NEG_INF_MASK;
            if (kbase + 16 + nn > row) s1 = NEG_INF_MASK;
            const float rm = redmax16(fmaxf(s0, s1));
            const float mn = fmaxf(mrow[r], rm);
            const float rs = redsum16(__expf(s0 - mn) + __expf(s1 - mn));
            lrow[r] = lrow[r] * __expf(mrow[r] - mn) + rs;
            mrow[r] = mn;
        }
    }

    float invl[8];
#pragma unroll
    for (int r = 0; r < 8; ++r) invl[r] = 1.0f / lrow[r];

    // ---- pass 2: normalized probabilities out + O = P*V ----
    v8f o0 = {}, o1 = {}, o2 = {}, o3 = {};
    for (int ch = 0; ch < nchunks; ++ch) {
        const int kbase = ch * 32;
        __syncthreads();
        stage32x64_blk(kp + kbase * 64, lds_k, tid);
        stage32x64_blk(vp + kbase * 64, lds_v, tid);
        __syncthreads();
        if (ch + 1 < nchunks) {
            __builtin_prefetch(kp + (kbase + 32) * 64 + tid * 16, 0, 1);
            __builtin_prefetch(vp + (kbase + 32) * 64 + tid * 16, 0, 1);
        }

        v8f t0, t1;
        score_chunk(aq0, aq1, lds_k, lane, t0, t1);

        // Hoist V B-fragments: independent of P, hidden under softmax VALU.
        FragBF bv0, bv1, bv2, bv3;
        load_bfrag_v(lds_v, 0, lane, bv0);
        load_bfrag_v(lds_v, 16, lane, bv1);
        load_bfrag_v(lds_v, 32, lane, bv2);
        load_bfrag_v(lds_v, 48, lane, bv3);

#pragma unroll
        for (int r = 0; r < 8; ++r) {
            const int row = q0 + r + 8 * hfl;
            float s0 = t0[r] * SCALE, s1 = t1[r] * SCALE;
            if (kbase + nn > row)      s0 = NEG_INF_MASK;
            if (kbase + 16 + nn > row) s1 = NEG_INF_MASK;
            const float p0 = __expf(s0 - mrow[r]) * invl[r];
            const float p1 = __expf(s1 - mrow[r]) * invl[r];
            ap[(long)row * S + kbase + nn]      = p0;
            ap[(long)row * S + kbase + 16 + nn] = p1;
            const int prow = r + 8 * hfl;
            myp[prow * 32 + nn]      = f32_to_bf16(p0);
            myp[prow * 32 + 16 + nn] = f32_to_bf16(p1);
        }
        __syncthreads();   // P tile visible -> reshape to A-fragment

        FragBF apf;
        load_afrag(myp, 32, lane, apf);
        o0 = __builtin_amdgcn_wmma_f32_16x16x32_bf16(false, apf.v, false, bv0.v,
                                                     (short)0, o0, false, false);
        o1 = __builtin_amdgcn_wmma_f32_16x16x32_bf16(false, apf.v, false, bv1.v,
                                                     (short)0, o1, false, false);
        o2 = __builtin_amdgcn_wmma_f32_16x16x32_bf16(false, apf.v, false, bv2.v,
                                                     (short)0, o2, false, false);
        o3 = __builtin_amdgcn_wmma_f32_16x16x32_bf16(false, apf.v, false, bv3.v,
                                                     (short)0, o3, false, false);
    }

    // ---- write O tile (16x64 per wave) ----
#pragma unroll
    for (int r = 0; r < 8; ++r) {
        const long row = q0 + r + 8 * hfl;
        op[row * D + 0  + nn] = o0[r];
        op[row * D + 16 + nn] = o1[r];
        op[row * D + 32 + nn] = o2[r];
        op[row * D + 48 + nn] = o3[r];
    }

    // ---- zero-fill fully-masked keys [kend, S) for rows [base, base+64) ----
    for (int r = 0; r < 64; ++r) {
        float4* rowp = (float4*)(ap + (long)(base + r) * S);
        for (int c4 = kend / 4 + tid; c4 < S / 4; c4 += 128)
            rowp[c4] = make_float4(0.f, 0.f, 0.f, 0.f);
    }
}

extern "C" void kernel_launch(void* const* d_in, const int* in_sizes, int n_in,
                              void* d_out, int out_size, void* d_ws, size_t ws_size,
                              hipStream_t stream) {
    (void)in_sizes; (void)n_in; (void)out_size; (void)d_ws; (void)ws_size;
    const float* q = (const float*)d_in[0];
    const float* k = (const float*)d_in[1];
    const float* v = (const float*)d_in[2];
    // d_in[3] is the causal mask; causality is computed analytically.
    float* out  = (float*)d_out;
    float* attn = out + (long)B * H * S * D;   // outputs concatenated: out, attn
    dim3 grid(S / 64, H, B), block(128);
    hipLaunchKernelGGL(fa_fwd_kernel, grid, block, 0, stream, q, k, v, out, attn);
}
